// GraphConv_Ortega_36404142801454
// MI455X (gfx1250) — matrix operand order repaired and checked by reading, not compiled, not run on hardware
//
#include <hip/hip_runtime.h>

// ---------------------------------------------------------------------------
// GraphConv_Ortega for MI455X (gfx1250):
//   deg/L build -> parallel Jacobi eigensolver (order/sign-invariant use of U)
//   -> bf16 WMMA GEMMs:
//        agg  = U^T X          (LDS-staged, async double-buffered)
//        h    = relu(agg W1+b1) (direct, K small / weights cache-resident)
//        midT = (h W2 + b2)^T   (direct, transposed store per batch)
//        out  = U mid           (LDS-staged, async double-buffered, f32 out)
// ---------------------------------------------------------------------------

typedef __attribute__((ext_vector_type(16))) __bf16 v16bf;
typedef __attribute__((ext_vector_type(8)))  float  v8f;
typedef __attribute__((ext_vector_type(4)))  unsigned u32x4;
typedef int gv4i __attribute__((vector_size(16)));   // matches async-LDS builtin param

#define AS1 __attribute__((address_space(1)))
#define AS3 __attribute__((address_space(3)))

#if __has_builtin(__builtin_amdgcn_global_load_async_to_lds_b128)
#define HAS_ASYNC_LDS 1
#endif

union Frag { unsigned u[8]; v16bf v; };

__device__ __forceinline__ unsigned short f2bf(float f) {
  unsigned x = __float_as_uint(f);
  x += 0x7FFFu + ((x >> 16) & 1u);            // round-to-nearest-even
  return (unsigned short)(x >> 16);
}

__device__ __forceinline__ void wait_async_zero() {
#if __has_builtin(__builtin_amdgcn_s_wait_asynccnt)
  __builtin_amdgcn_s_wait_asynccnt(0);
#elif defined(HAS_ASYNC_LDS)
  asm volatile("s_wait_asynccnt 0x0" ::: "memory");
#endif
}

// -------------------- degree + normalized Laplacian ------------------------

__global__ void deg_kernel(const float* __restrict__ A, float* __restrict__ deg,
                           float* __restrict__ dinv, int n) {
  int j = blockIdx.x * blockDim.x + threadIdx.x;
  if (j >= n) return;
  float s = 0.f;
  for (int i = 0; i < n; ++i) s += A[(size_t)i * n + j];   // column sum
  for (int k = 0; k < n; ++k) s += A[(size_t)j * n + k];   // row sum
  s -= A[(size_t)j * n + j];
  deg[j]  = s;
  dinv[j] = (s > 0.f) ? rsqrtf(s) : 0.f;
}

__global__ void build_l_kernel(const float* __restrict__ A, const float* __restrict__ deg,
                               const float* __restrict__ dinv,
                               float* __restrict__ L, float* __restrict__ U, int n) {
  size_t idx = (size_t)blockIdx.x * blockDim.x + threadIdx.x;
  if (idx >= (size_t)n * n) return;
  int i = (int)(idx / n), j = (int)(idx % n);
  float l = ((i == j) ? deg[i] : 0.f) - A[idx];
  L[idx] = dinv[i] * l * dinv[j];
  U[idx] = (i == j) ? 1.f : 0.f;
}

// -------------------- parallel cyclic Jacobi eigensolver -------------------

__global__ __launch_bounds__(1024) void jacobi_kernel(float* __restrict__ L,
                                                      float* __restrict__ U,
                                                      int nsweeps) {
  const int n = 1024, half = 512;
  __shared__ float sc[half], ss[half];
  __shared__ int   sp[half], sq[half];
  const int tid = threadIdx.x;

  for (int sweep = 0; sweep < nsweeps; ++sweep) {
    for (int step = 0; step < n - 1; ++step) {
      if (tid < half) {
        int a = (tid == 0) ? 0 : 1 + (tid - 1 + step) % (n - 1);
        int b = 1 + (n - 2 - tid + step) % (n - 1);
        int p = min(a, b), q = max(a, b);
        float app = L[(size_t)p * n + p];
        float aqq = L[(size_t)q * n + q];
        float apq = L[(size_t)p * n + q];
        float c = 1.f, s = 0.f;
        if (fabsf(apq) > 1e-10f) {
          float tau = (aqq - app) / (2.f * apq);
          float t = ((tau >= 0.f) ? 1.f : -1.f) / (fabsf(tau) + sqrtf(1.f + tau * tau));
          c = rsqrtf(1.f + t * t);
          s = t * c;
        }
        sc[tid] = c; ss[tid] = s; sp[tid] = p; sq[tid] = q;
      }
      __syncthreads();
      for (int idx = tid; idx < half * n; idx += 1024) {   // L <- J^T L
        int pair = idx >> 10, k = idx & (n - 1);
        int p = sp[pair], q = sq[pair];
        float c = sc[pair], s = ss[pair];
        float lp = L[(size_t)p * n + k], lq = L[(size_t)q * n + k];
        L[(size_t)p * n + k] = c * lp - s * lq;
        L[(size_t)q * n + k] = s * lp + c * lq;
      }
      __syncthreads();
      for (int idx = tid; idx < half * n; idx += 1024) {   // L <- L J ; U <- U J
        int pair = idx >> 10, k = idx & (n - 1);
        int p = sp[pair], q = sq[pair];
        float c = sc[pair], s = ss[pair];
        float lp = L[(size_t)k * n + p], lq = L[(size_t)k * n + q];
        L[(size_t)k * n + p] = c * lp - s * lq;
        L[(size_t)k * n + q] = s * lp + c * lq;
        float up = U[(size_t)k * n + p], uq = U[(size_t)k * n + q];
        U[(size_t)k * n + p] = c * up - s * uq;
        U[(size_t)k * n + q] = s * up + c * uq;
      }
      __syncthreads();
    }
  }
}

// -------------------- layout conversion kernels ----------------------------

__global__ void conv_bf16_kernel(const float* __restrict__ src,
                                 unsigned short* __restrict__ dst, size_t count) {
  size_t i = (size_t)blockIdx.x * blockDim.x + threadIdx.x;
  if (i < count) dst[i] = f2bf(src[i]);
}

__global__ void btrans_bf16_kernel(const float* __restrict__ src,
                                   unsigned short* __restrict__ dst,
                                   int rows, int cols) {
  size_t bo = (size_t)blockIdx.y * rows * cols;
  size_t i  = (size_t)blockIdx.x * blockDim.x + threadIdx.x;
  if (i >= (size_t)rows * cols) return;
  int r = (int)(i / cols), c = (int)(i % cols);
  dst[bo + (size_t)c * rows + r] = f2bf(src[bo + i]);
}

// -------------------- direct-load bf16 WMMA GEMM (small K / small B) -------

template <bool RELU, int OUTM>   // OUTM: 0 bf16 row-major, 1 bf16 transposed-per-batch
__global__ __launch_bounds__(256)
void gemm_wmma_kernel(const unsigned short* __restrict__ Abf,
                      const unsigned short* __restrict__ Btbf,
                      const float* __restrict__ bias,
                      void* __restrict__ Cv,
                      int M, int Nc, int K, int batN) {
  const int lane = threadIdx.x & 31;
  const int lr   = lane & 15;
  const bool hi  = lane >= 16;
  const int wave = blockIdx.x * (blockDim.x >> 5) + (threadIdx.x >> 5);
  const int tilesN   = Nc >> 6;
  const int numTiles = (M >> 4) * tilesN;
  if (wave >= numTiles) return;                  // wave-uniform
  const int tm = wave / tilesN;
  const int tn = wave % tilesN;

  const unsigned* arow = (const unsigned*)(Abf + (size_t)(tm * 16 + lr) * K);
  const unsigned* brow[4];
#pragma unroll
  for (int j = 0; j < 4; ++j)
    brow[j] = (const unsigned*)(Btbf + (size_t)(tn * 64 + j * 16 + lr) * K);

  v8f z = {};
  v8f acc[4] = {z, z, z, z};
  const int aShift = hi ? 8 : 0;
  const int bShift = hi ? 8 : 0;

  for (int k0 = 0; k0 < K; k0 += 32) {
    const int ad = (k0 >> 1) + aShift;
    const int bd = (k0 >> 1) + bShift;
    Frag a;
#pragma unroll
    for (int v = 0; v < 8; ++v) a.u[v] = arow[ad + v];
    __builtin_prefetch(arow + ad + 16, 0, 1);
    Frag b[4];
#pragma unroll
    for (int j = 0; j < 4; ++j) {
#pragma unroll
      for (int v = 0; v < 8; ++v) b[j].u[v] = brow[j][bd + v];
    }
#pragma unroll
    for (int j = 0; j < 4; ++j)
      acc[j] = __builtin_amdgcn_wmma_f32_16x16x32_bf16(
          false, a.v, false, b[j].v, (short)0, acc[j], false, false);
  }

  const int mbase = tm * 16 + (hi ? 8 : 0);
#pragma unroll
  for (int j = 0; j < 4; ++j) {
    const int ncol = tn * 64 + j * 16 + lr;
    const float bv = bias ? bias[ncol] : 0.f;
#pragma unroll
    for (int r = 0; r < 8; ++r) {
      const int m = mbase + r;
      float val = acc[j][r] + bv;
      if (RELU) val = fmaxf(val, 0.f);
      if constexpr (OUTM == 0) {
        ((unsigned short*)Cv)[(size_t)m * Nc + ncol] = f2bf(val);
      } else {
        const int bb = m / batN, nn = m % batN;
        ((unsigned short*)Cv)[((size_t)bb * Nc + ncol) * batN + nn] = f2bf(val);
      }
    }
  }
}

// -------------------- LDS-staged, async double-buffered WMMA GEMM ----------
// Macro tile 128(M) x 64(N) per 256-thread block (8 waves, each 16x64).
// Per k-step (32): stage A(128x32) + B(64x32) bf16 into LDS with per-lane
// 16B async global->LDS copies (ASYNCcnt), double buffered so the DMA of
// stage s+1 overlaps ds_load+WMMA of stage s.

constexpr int ADW = 128 * 16;            // A-tile dwords
constexpr int STAGE_DW = ADW + 64 * 16;  // + B-tile dwords (3072 dwords = 12KB)

__device__ __forceinline__ void async_cp16(const unsigned short* src, unsigned* dst) {
#ifdef HAS_ASYNC_LDS
  __builtin_amdgcn_global_load_async_to_lds_b128((AS1 gv4i*)src, (AS3 gv4i*)dst, 0, 0);
#else
  *(u32x4*)dst = *(const u32x4*)src;
#endif
}

__device__ __forceinline__ void stage_tile(const unsigned short* __restrict__ A,
                                           const unsigned short* __restrict__ Bt,
                                           unsigned* __restrict__ ldsbuf,
                                           int k0, int K, int tid) {
#pragma unroll
  for (int i = 0; i < 2; ++i) {          // A: 512 x 16B chunks
    const int c = tid + i * 256;
    const int r = c >> 2, part = c & 3;
    async_cp16(A + (size_t)r * K + k0 + part * 8, ldsbuf + r * 16 + part * 4);
  }
  {                                      // B: 256 x 16B chunks
    const int r = tid >> 2, part = tid & 3;
    async_cp16(Bt + (size_t)r * K + k0 + part * 8, ldsbuf + ADW + r * 16 + part * 4);
  }
}

template <int OUTM>   // 0 = bf16 row-major, 2 = f32 row-major
__global__ __launch_bounds__(256)
void gemm_wmma_lds_kernel(const unsigned short* __restrict__ Abf,
                          const unsigned short* __restrict__ Btbf,
                          void* __restrict__ Cv,
                          int M, int Nc, int K,
                          size_t aStride, size_t btStride, size_t cStride) {
  __shared__ unsigned lds[2][STAGE_DW];
  const int tid  = threadIdx.x;
  const int lane = tid & 31, lr = lane & 15;
  const bool hi  = lane >= 16;
  const int wave = tid >> 5;
  const int nBlk64 = Nc >> 6;
  const int mBlk = blockIdx.x / nBlk64;
  const int nBlk = blockIdx.x % nBlk64;
  const int batch = blockIdx.y;

  const unsigned short* A  = Abf  + (size_t)batch * aStride  + (size_t)(mBlk * 128) * K;
  const unsigned short* Bt = Btbf + (size_t)batch * btStride + (size_t)(nBlk * 64) * K;

  v8f z = {};
  v8f acc[4] = {z, z, z, z};
  const int aOff = (wave * 16 + lr) * 16 + (hi ? 8 : 0);
  const int bShift = hi ? 8 : 0;
  const int nsteps = K >> 5;

  stage_tile(A, Bt, lds[0], 0, K, tid);          // prologue: stage 0

  for (int s = 0; s < nsteps; ++s) {
    const int buf = s & 1;
    wait_async_zero();                           // stage s landed in LDS
    __syncthreads();                             // visible to all 8 waves
    if (s + 1 < nsteps)
      stage_tile(A, Bt, lds[buf ^ 1], (s + 1) * 32, K, tid);  // overlap DMA

    const unsigned* la = &lds[buf][aOff];
    Frag a;
#pragma unroll
    for (int v = 0; v < 8; ++v) a.u[v] = la[v];
    Frag b[4];
#pragma unroll
    for (int j = 0; j < 4; ++j) {
      const unsigned* lb = &lds[buf][ADW + (j * 16 + lr) * 16 + bShift];
#pragma unroll
      for (int v = 0; v < 8; ++v) b[j].u[v] = lb[v];
    }
#pragma unroll
    for (int j = 0; j < 4; ++j)
      acc[j] = __builtin_amdgcn_wmma_f32_16x16x32_bf16(
          false, a.v, false, b[j].v, (short)0, acc[j], false, false);
    __syncthreads();                             // done reading buf before reuse
  }

  const int mbase = mBlk * 128 + wave * 16 + (hi ? 8 : 0);
#pragma unroll
  for (int j = 0; j < 4; ++j) {
    const int ncol = nBlk * 64 + j * 16 + lr;
#pragma unroll
    for (int r = 0; r < 8; ++r) {
      const int m = mbase + r;
      const float val = acc[j][r];
      if constexpr (OUTM == 0) {
        ((unsigned short*)Cv)[(size_t)batch * cStride + (size_t)m * Nc + ncol] = f2bf(val);
      } else {
        ((float*)Cv)[(size_t)batch * cStride + (size_t)m * Nc + ncol] = val;
      }
    }
  }
}

// -------------------- host-side orchestration ------------------------------

extern "C" void kernel_launch(void* const* d_in, const int* in_sizes, int n_in,
                              void* d_out, int out_size, void* d_ws, size_t ws_size,
                              hipStream_t stream) {
  const int N = 1024, Bn = 64, DIN = 256, HID = 128, DOUT = 256;

  const float* feat = (const float*)d_in[0];   // [B, N, DIN]
  const float* Aadj = (const float*)d_in[1];   // [N, N]
  const float* W1   = (const float*)d_in[2];   // [DIN, HID]
  const float* b1   = (const float*)d_in[3];   // [HID]
  const float* W2   = (const float*)d_in[4];   // [HID, DOUT]
  const float* b2   = (const float*)d_in[5];   // [DOUT]
  float* out = (float*)d_out;                  // [B, N, DOUT]

  char* ws = (char*)d_ws;
  size_t off = 0;
  auto alloc = [&](size_t bytes) -> void* {
    void* p = ws + off;
    off = (off + bytes + 255) & ~(size_t)255;
    return p;
  };

  float* L    = (float*)alloc((size_t)N * N * 4);
  float* U    = (float*)alloc((size_t)N * N * 4);
  float* deg  = (float*)alloc((size_t)N * 4);
  float* dinv = (float*)alloc((size_t)N * 4);
  unsigned short* Ubf  = (unsigned short*)alloc((size_t)N * N * 2);        // U   [N x K=N]
  unsigned short* Utbf = (unsigned short*)alloc((size_t)N * N * 2);        // U^T [N x K=N]
  unsigned short* W1t  = (unsigned short*)alloc((size_t)DIN * HID * 2);    // [HID x DIN]
  unsigned short* W2t  = (unsigned short*)alloc((size_t)HID * DOUT * 2);   // [DOUT x HID]
  unsigned short* Xt   = (unsigned short*)alloc((size_t)Bn * DIN * N * 2); // X^T / midT alias
  unsigned short* agg  = (unsigned short*)alloc((size_t)Bn * N * DIN * 2);
  unsigned short* hbuf = (unsigned short*)alloc((size_t)Bn * N * HID * 2);
  unsigned short* midT = Xt;  // Xt dead after GEMM1; reuse for (h W2 + b2)^T

  // 1) degrees + normalized Laplacian (+ U := I)
  deg_kernel<<<dim3((N + 255) / 256), 256, 0, stream>>>(Aadj, deg, dinv, N);
  build_l_kernel<<<dim3((N * N) / 256), 256, 0, stream>>>(Aadj, deg, dinv, L, U, N);

  // 2) Jacobi eigensolver (10 full cyclic sweeps)
  jacobi_kernel<<<1, 1024, 0, stream>>>(L, U, 10);

  // 3) precision/layout conversion for WMMA operands
  conv_bf16_kernel<<<dim3((N * N) / 256), 256, 0, stream>>>(U, Ubf, (size_t)N * N);
  btrans_bf16_kernel<<<dim3((N * N) / 256, 1), 256, 0, stream>>>(U, Utbf, N, N);
  btrans_bf16_kernel<<<dim3((DIN * HID) / 256, 1), 256, 0, stream>>>(W1, W1t, DIN, HID);
  btrans_bf16_kernel<<<dim3((HID * DOUT) / 256, 1), 256, 0, stream>>>(W2, W2t, HID, DOUT);
  btrans_bf16_kernel<<<dim3((N * DIN) / 256, Bn), 256, 0, stream>>>(feat, Xt, N, DIN);

  // 4) agg_b = U^T X_b   (LDS-staged; A = U^T shared across batches, Bt = X_b^T)
  {
    const int M = N, Nc = DIN, K = N;
    const int blocks = (M / 128) * (Nc / 64);      // 32 per batch
    gemm_wmma_lds_kernel<0><<<dim3(blocks, Bn), 256, 0, stream>>>(
        Utbf, Xt, agg, M, Nc, K,
        (size_t)0, (size_t)DIN * N, (size_t)N * DIN);
  }
  // 5) h = relu(agg W1 + b1)   (direct: K=256, W1t is 64KB -> cache resident)
  {
    const int M = Bn * N, Nc = HID, K = DIN;
    const int tiles = (M / 16) * (Nc / 64);
    gemm_wmma_kernel<true, 0><<<dim3((tiles + 7) / 8, 1), 256, 0, stream>>>(
        agg, W1t, b1, hbuf, M, Nc, K, 0);
  }
  // 6) midT_b = (h_b W2 + b2)^T  stored [DOUT x N] per batch -> GEMM4's Bt
  {
    const int M = Bn * N, Nc = DOUT, K = HID;
    const int tiles = (M / 16) * (Nc / 64);
    gemm_wmma_kernel<false, 1><<<dim3((tiles + 7) / 8, 1), 256, 0, stream>>>(
        hbuf, W2t, b2, midT, M, Nc, K, N);
  }
  // 7) out_b = U mid_b   (LDS-staged, f32 output straight to d_out)
  {
    const int M = N, Nc = DOUT, K = N;
    const int blocks = (M / 128) * (Nc / 64);
    gemm_wmma_lds_kernel<2><<<dim3(blocks, Bn), 256, 0, stream>>>(
        Ubf, midT, out, M, Nc, K,
        (size_t)0, (size_t)DOUT * N, (size_t)N * DOUT);
  }
}
